// Model_16655883174575
// MI455X (gfx1250) — compile-verified, matrix operand order checked by reading, not run on hardware
//
#include <hip/hip_runtime.h>
#include <cstdint>
#include <cstddef>

// ---------------------------------------------------------------------------
// Types / constants
// ---------------------------------------------------------------------------
typedef __attribute__((ext_vector_type(16))) _Float16 v16h;
typedef __attribute__((ext_vector_type(8)))  _Float16 v8h;
typedef __attribute__((ext_vector_type(8)))  float    v8f;

constexpr int Bc = 32;     // batch
constexpr int Tc = 4096;   // time
constexpr int Hc = 32;     // hidden
constexpr int Gc = 128;    // 4*H gates

#define DEVINL __device__ __forceinline__

// Fast transcendentals: v_exp_f32 + v_rcp_f32, no IEEE division expansion.
DEVINL float sigmoidf_(float x) { return __builtin_amdgcn_rcpf(1.0f + __expf(-x)); }
DEVINL float tanhf_(float x) {
  return __builtin_fmaf(-2.0f, __builtin_amdgcn_rcpf(1.0f + __expf(2.0f * x)), 1.0f);
}

DEVINL v16h pack16(v8h lo, v8h hi) {
  return __builtin_shufflevector(lo, hi, 0, 1, 2, 3, 4, 5, 6, 7, 8, 9, 10, 11, 12, 13, 14, 15);
}

DEVINL v8f wmma_f16(v16h a, v16h b, v8f c) {
  return __builtin_amdgcn_wmma_f32_16x16x32_f16(false, a, false, b, (short)0, c, false, false);
}

// ---------------------------------------------------------------------------
// K1: x_prim = mask * (x * ffw + ffb), f32 + f16 copies, layout (B,T,H)
// ---------------------------------------------------------------------------
__global__ void k_ffn(const float* __restrict__ x, const int* __restrict__ len,
                      const float* __restrict__ ffw, const float* __restrict__ ffb,
                      float* __restrict__ xp, _Float16* __restrict__ xph) {
  int idx = blockIdx.x * blockDim.x + threadIdx.x;
  if (idx >= Bc * Tc * Hc) return;
  int h = idx & (Hc - 1);
  int t = (idx >> 5) & (Tc - 1);
  int b = idx >> 17;
  float v = (t < len[b]) ? x[b * Tc + t] * ffw[h] + ffb[h] : 0.0f;
  xp[idx] = v;
  xph[idx] = (_Float16)v;
}

// ---------------------------------------------------------------------------
// K2: input-gate GEMM  out[r,g] = sum_k in[r,k]*W[g,k] + (bih[g]+bhh[g])
//     R = B*T rows, 128 gate cols, DIN = 32 or 64.  WMMA f16 -> f32.
// ---------------------------------------------------------------------------
template <int DIN>
__global__ void __launch_bounds__(256)
k_gates_gemm(const _Float16* __restrict__ in, const float* __restrict__ w,
             const float* __restrict__ bih, const float* __restrict__ bhh,
             float* __restrict__ out) {
  __shared__ __attribute__((aligned(16))) _Float16 wl[Gc][DIN + 8];  // +8: bank spread
  __shared__ float bl[Gc];
  for (int i = threadIdx.x; i < Gc * DIN; i += blockDim.x)
    wl[i / DIN][i % DIN] = (_Float16)w[i];
  for (int i = threadIdx.x; i < Gc; i += blockDim.x)
    bl[i] = bih[i] + bhh[i];
  __syncthreads();

  const int lane = threadIdx.x & 31;
  const int wave = threadIdx.x >> 5;
  const int nwav = blockDim.x >> 5;
  const int n = lane & 15;
  const int koff = (lane < 16) ? 0 : 8;   // f16 operand K-split per half-wave
  constexpr int KC = DIN / 32;
  const long RT = (long)Bc * Tc / 16;

  for (long rt = (long)blockIdx.x * nwav + wave; rt < RT; rt += (long)gridDim.x * nwav) {
    v16h a[KC];
    const _Float16* arow = in + (rt * 16 + n) * DIN;
#pragma unroll
    for (int kc = 0; kc < KC; ++kc) {
      v8h lo = *(const v8h*)(arow + kc * 32 + koff);
      v8h hi = *(const v8h*)(arow + kc * 32 + koff + 16);
      a[kc] = pack16(lo, hi);
    }
#pragma unroll
    for (int ct = 0; ct < 8; ++ct) {
      float bv = bl[ct * 16 + n];
      v8f acc = {bv, bv, bv, bv, bv, bv, bv, bv};
#pragma unroll
      for (int kc = 0; kc < KC; ++kc) {
        const _Float16* brow = &wl[ct * 16 + n][kc * 32 + koff];
        v16h bm = pack16(*(const v8h*)brow, *(const v8h*)(brow + 16));
        acc = wmma_f16(a[kc], bm, acc);
      }
      float* orow = out + rt * 16 * Gc + ct * 16 + n;
#pragma unroll
      for (int v = 0; v < 8; ++v) {
        int mm = (lane < 16) ? v : v + 8;
        orow[(long)mm * Gc] = acc[v];
      }
    }
  }
}

// ---------------------------------------------------------------------------
// K3: LSTM recurrence, one block (2 waves) per direction; grid.x = 2.
//     Per-row lengths & row offsets cached in registers; branchless per-step
//     address generation; 64 clauseable gathers + 16 next-step prefetches,
//     then 8 WMMAs per wave/step.
// ---------------------------------------------------------------------------
__global__ void __launch_bounds__(64)
k_recur(const float* __restrict__ xgF, const float* __restrict__ xgR,
        const float* __restrict__ whhF, const float* __restrict__ whhR,
        const int* __restrict__ len,
        float* __restrict__ hseq, _Float16* __restrict__ hseqh) {
  const int dir = blockIdx.x;
  const float* __restrict__ xg = dir ? xgR : xgF;
  const float* __restrict__ whh = dir ? whhR : whhF;

  __shared__ __attribute__((aligned(16))) _Float16 wl[Gc][Hc + 8];
  __shared__ __attribute__((aligned(16))) _Float16 hl[Bc][Hc + 8];
  __shared__ int ll[Bc];

  for (int i = threadIdx.x; i < Gc * Hc; i += blockDim.x) wl[i >> 5][i & 31] = (_Float16)whh[i];
  for (int i = threadIdx.x; i < Bc * Hc; i += blockDim.x) hl[i >> 5][i & 31] = (_Float16)0.0f;
  for (int i = threadIdx.x; i < Bc; i += blockDim.x) ll[i] = len[i];
  __syncthreads();

  const int wave = threadIdx.x >> 5;
  const int lane = threadIdx.x & 31;
  const int n = lane & 15;
  const int koff = (lane < 16) ? 0 : 8;
  const int hid = wave * 16 + n;        // this wave owns hidden cols [16w, 16w+16)
  const int gbase = wave * 16 + n;      // col within each 32-wide gate block
  const int dstep = dir ? -Gc : Gc;     // next-step stride through the gate buffer

  // B operands for the 4 gates (gate j -> cols j*32 + wave*16 + n)
  v16h bm[4];
#pragma unroll
  for (int j = 0; j < 4; ++j) {
    const _Float16* brow = &wl[j * 32 + wave * 16 + n][koff];
    bm[j] = pack16(*(const v8h*)brow, *(const v8h*)(brow + 16));
  }

  // Register-cached per-row metadata (row bb handled by (tr,v) of this lane)
  int Lr[16], rowTG[16];
#pragma unroll
  for (int tr = 0; tr < 2; ++tr)
#pragma unroll
    for (int v = 0; v < 8; ++v) {
      int bb = ((lane < 16) ? v : v + 8) + tr * 16;
      Lr[tr * 8 + v] = ll[bb];
      rowTG[tr * 8 + v] = bb * Tc * Gc;  // 16.7M max, fits int
    }

  v8f cst0 = {}, cst1 = {};  // c state (C layout), batch rows 0..15 / 16..31

  for (int s = 0; s < Tc; ++s) {
    // ---- A operands (H matrix) from LDS, 16-bit A layout ----
    v16h a0 = pack16(*(const v8h*)&hl[n][koff],      *(const v8h*)&hl[n][koff + 16]);
    v16h a1 = pack16(*(const v8h*)&hl[n + 16][koff], *(const v8h*)&hl[n + 16][koff + 16]);
    __syncthreads();  // hl reads complete before anyone rewrites it

    // ---- per-row gate-buffer offsets for this step (branchless) ----
    int off[16];
#pragma unroll
    for (int u = 0; u < 16; ++u) {
      int tt = dir ? (Lr[u] - 1 - s) : s;
      tt = tt < 0 ? 0 : tt;
      off[u] = rowTG[u] + tt * Gc + gbase;
    }

    // ---- gather gate pre-activations (C operands): 4 imm-offset loads/row,
    //      plus a prefetch of next step's row (address is ptr +/- Gc) ----
    v8f acc[4][2];
#pragma unroll
    for (int tr = 0; tr < 2; ++tr)
#pragma unroll
      for (int v = 0; v < 8; ++v) {
        const float* p = xg + off[tr * 8 + v];
#pragma unroll
        for (int j = 0; j < 4; ++j)
          acc[j][tr][v] = p[j * 32];
        __builtin_prefetch(p + dstep, 0, 1);  // global_prefetch_b8, next time step
      }

    // ---- gates += H @ W_hh^T : 8 WMMAs, accumulate in place ----
#pragma unroll
    for (int j = 0; j < 4; ++j) {
      acc[j][0] = wmma_f16(a0, bm[j], acc[j][0]);
      acc[j][1] = wmma_f16(a1, bm[j], acc[j][1]);
    }

    // ---- elementwise LSTM cell + masked state/output update ----
#pragma unroll
    for (int tr = 0; tr < 2; ++tr)
#pragma unroll
      for (int v = 0; v < 8; ++v) {
        int u = tr * 8 + v;
        int bb = ((lane < 16) ? v : v + 8) + tr * 16;
        float cprev = tr ? cst1[v] : cst0[v];
        float ig = sigmoidf_(acc[0][tr][v]);
        float fg = sigmoidf_(acc[1][tr][v]);
        float gg = tanhf_(acc[2][tr][v]);
        float og = sigmoidf_(acc[3][tr][v]);
        float cn = fg * cprev + ig * gg;
        float hn = og * tanhf_(cn);
        if (s < Lr[u]) {
          if (tr) cst1[v] = cn; else cst0[v] = cn;
          // (bb*T + tt)*64 = (off - gbase) >> 1   (off is (bb*T+tt)*128 + gbase)
          long o = ((long)(off[u] - gbase) >> 1) + dir * 32 + hid;
          if (hseq)  hseq[o]  = hn;
          if (hseqh) hseqh[o] = (_Float16)hn;
          hl[bb][hid] = (_Float16)hn;
        }
      }
    __syncthreads();  // hl writes visible before next step's reads
  }
}

// ---------------------------------------------------------------------------
// K4: masked mean / max / last pooling over time.  grid=B, block=64.
// ---------------------------------------------------------------------------
__global__ void k_pool(const float* __restrict__ h1, const int* __restrict__ len,
                       float* __restrict__ hp) {
  int b = blockIdx.x;
  int c = threadIdx.x;  // 0..63
  int L = len[b];
  float s = 0.0f, mx = -1e30f;
  for (int t = 0; t < L; ++t) {
    float v = h1[((long)b * Tc + t) * 64 + c];
    s += v;
    mx = fmaxf(mx, v);
  }
  float last = h1[((long)b * Tc + (L - 1)) * 64 + c];
  hp[b * 192 + c] = s / (float)L;
  hp[b * 192 + 64 + c] = mx;
  hp[b * 192 + 128 + c] = last;
}

// ---------------------------------------------------------------------------
// K5: conv1 (1 -> 64 ch, stride 2, pad 1, k=2x2) from f32 x_prim, f16 out+ReLU
// ---------------------------------------------------------------------------
__global__ void k_conv1(const float* __restrict__ xp, const float* __restrict__ w,
                        const float* __restrict__ bias, _Float16* __restrict__ out) {
  const int OH = 2049, OW = 17, CO = 64;
  long idx = (long)blockIdx.x * blockDim.x + threadIdx.x;
  long total = (long)Bc * CO * OH * OW;
  if (idx >= total) return;
  int ox = (int)(idx % OW);
  long r = idx / OW;
  int oy = (int)(r % OH); r /= OH;
  int co = (int)(r % CO);
  int b = (int)(r / CO);
  float acc = bias[co];
#pragma unroll
  for (int ky = 0; ky < 2; ++ky) {
    int iy = 2 * oy - 1 + ky;
    if (iy < 0 || iy >= Tc) continue;
#pragma unroll
    for (int kx = 0; kx < 2; ++kx) {
      int ix = 2 * ox - 1 + kx;
      if (ix < 0 || ix >= Hc) continue;
      acc += w[co * 4 + ky * 2 + kx] * xp[((long)b * Tc + iy) * Hc + ix];
    }
  }
  out[idx] = (_Float16)fmaxf(acc, 0.0f);
}

// ---------------------------------------------------------------------------
// K6: WMMA implicit-GEMM conv (stride 2, pad 1, 2x2), f16 in/out, f32 acc.
//     Per block: one (batch, 16-position tile). im2col panel (16 x 4*CI f16)
//     staged once in LDS, reused by every output channel. A = weights
//     (float4 loads, f32 -> f16 on the fly), B = LDS panel, ReLU on store.
// ---------------------------------------------------------------------------
template <int CI>
__global__ void __launch_bounds__(256)
k_conv_wmma(const _Float16* __restrict__ in, const float* __restrict__ w,
            const float* __restrict__ bias, _Float16* __restrict__ out,
            int IH, int IW, int CO, int OH, int OW) {
  constexpr int K = CI * 4;
  __shared__ __attribute__((aligned(16))) _Float16 panel[16][K + 8];  // +8: bank spread

  const int S = OH * OW;
  const int ntiles = (S + 15) >> 4;
  const int b = blockIdx.x / ntiles;
  const int pt = blockIdx.x % ntiles;

  // ---- im2col gather into LDS (each element read exactly once) ----
  for (int e = threadIdx.x; e < 16 * K; e += blockDim.x) {
    int nn = e / K, k = e % K;                 // K is a power of two
    int ci = k >> 2, ky = (k >> 1) & 1, kx = k & 1;
    int p = pt * 16 + nn; if (p >= S) p = S - 1;
    int oy = p / OW, ox = p % OW;
    int iy = 2 * oy - 1 + ky, ix = 2 * ox - 1 + kx;
    float v = 0.0f;
    if (iy >= 0 && iy < IH && ix >= 0 && ix < IW)
      v = (float)in[((long)(b * CI + ci) * IH + iy) * IW + ix];
    panel[nn][k] = (_Float16)v;
  }
  __syncthreads();

  const int lane = threadIdx.x & 31;
  const int wave = threadIdx.x >> 5;
  const int n = lane & 15;
  const int koff = (lane < 16) ? 0 : 8;
  const int mtiles = CO >> 4;
  const int p = pt * 16 + n;

  for (int mt = wave; mt < mtiles; mt += 8) {
    v8f acc;
#pragma unroll
    for (int v = 0; v < 8; ++v)
      acc[v] = bias[mt * 16 + ((lane < 16) ? v : v + 8)];

    const float* wrow = w + (long)(mt * 16 + n) * K;  // A row: co = mt*16 + (lane&15)
    for (int kc = 0; kc < K / 32; ++kc) {
      const float4* w4 = (const float4*)(wrow + kc * 32 + koff);       // 16B aligned
      float4 f0 = w4[0], f1 = w4[1], f2 = w4[4], f3 = w4[5];           // k, k+4, k+16, k+20
      v16h am;
      am[0]  = (_Float16)f0.x; am[1]  = (_Float16)f0.y; am[2]  = (_Float16)f0.z; am[3]  = (_Float16)f0.w;
      am[4]  = (_Float16)f1.x; am[5]  = (_Float16)f1.y; am[6]  = (_Float16)f1.z; am[7]  = (_Float16)f1.w;
      am[8]  = (_Float16)f2.x; am[9]  = (_Float16)f2.y; am[10] = (_Float16)f2.z; am[11] = (_Float16)f2.w;
      am[12] = (_Float16)f3.x; am[13] = (_Float16)f3.y; am[14] = (_Float16)f3.z; am[15] = (_Float16)f3.w;
      const _Float16* brow = &panel[n][kc * 32 + koff];
      v16h bmv = pack16(*(const v8h*)brow, *(const v8h*)(brow + 16));
      acc = wmma_f16(am, bmv, acc);
    }
    if (p < S) {
#pragma unroll
      for (int v = 0; v < 8; ++v) {
        int co = mt * 16 + ((lane < 16) ? v : v + 8);
        out[(long)(b * CO + co) * S + p] = (_Float16)fmaxf(acc[v], 0.0f);
      }
    }
  }
}

// ---------------------------------------------------------------------------
// K7: global mean over conv4 spatial (257x3) -> (B,64)
// ---------------------------------------------------------------------------
__global__ void k_cpool(const _Float16* __restrict__ c4, float* __restrict__ cp) {
  int idx = blockIdx.x * blockDim.x + threadIdx.x;
  if (idx >= Bc * 64) return;
  int co = idx & 63;
  int b = idx >> 6;
  const int S = 257 * 3;
  float s = 0.0f;
  const _Float16* p = c4 + (long)(b * 64 + co) * S;
  for (int i = 0; i < S; ++i) s += (float)p[i];
  cp[idx] = s / (float)S;
}

// ---------------------------------------------------------------------------
// K8: final linear  out[b,cls] = lin_b[cls] + lin_w[cls,:] . [cpool|mean|max|last]
// ---------------------------------------------------------------------------
__global__ void k_final(const float* __restrict__ cp, const float* __restrict__ hp,
                        const float* __restrict__ lw, const float* __restrict__ lb,
                        float* __restrict__ out) {
  int idx = threadIdx.x;
  if (idx >= Bc * 4) return;
  int cls = idx & 3;
  int b = idx >> 2;
  float acc = lb[cls];
  for (int j = 0; j < 64; ++j)  acc += lw[cls * 256 + j] * cp[b * 64 + j];
  for (int j = 0; j < 192; ++j) acc += lw[cls * 256 + 64 + j] * hp[b * 192 + j];
  out[b * 4 + cls] = acc;
}

// ---------------------------------------------------------------------------
// Host orchestration
// ---------------------------------------------------------------------------
static inline size_t aup(size_t x) { return (x + 255) & ~(size_t)255; }

extern "C" void kernel_launch(void* const* d_in, const int* in_sizes, int n_in,
                              void* d_out, int out_size, void* d_ws, size_t ws_size,
                              hipStream_t stream) {
  (void)in_sizes; (void)n_in; (void)out_size; (void)ws_size;
  const float* x       = (const float*)d_in[0];
  const int*   lengths = (const int*)d_in[1];
  const float* ffw     = (const float*)d_in[2];
  const float* ffb     = (const float*)d_in[3];
  const float* wih0f = (const float*)d_in[4],  *whh0f = (const float*)d_in[5];
  const float* bih0f = (const float*)d_in[6],  *bhh0f = (const float*)d_in[7];
  const float* wih0r = (const float*)d_in[8],  *whh0r = (const float*)d_in[9];
  const float* bih0r = (const float*)d_in[10], *bhh0r = (const float*)d_in[11];
  const float* wih1f = (const float*)d_in[12], *whh1f = (const float*)d_in[13];
  const float* bih1f = (const float*)d_in[14], *bhh1f = (const float*)d_in[15];
  const float* wih1r = (const float*)d_in[16], *whh1r = (const float*)d_in[17];
  const float* bih1r = (const float*)d_in[18], *bhh1r = (const float*)d_in[19];
  const float* c1w = (const float*)d_in[20], *c1b = (const float*)d_in[21];
  const float* c2w = (const float*)d_in[22], *c2b = (const float*)d_in[23];
  const float* c3w = (const float*)d_in[24], *c3b = (const float*)d_in[25];
  const float* c4w = (const float*)d_in[26], *c4b = (const float*)d_in[27];
  const float* lw  = (const float*)d_in[28], *lb  = (const float*)d_in[29];

  const size_t SZ_XP  = (size_t)Bc * Tc * Hc * sizeof(float);
  const size_t SZ_XPH = (size_t)Bc * Tc * Hc * sizeof(_Float16);
  const size_t SZ_XG  = (size_t)Bc * Tc * Gc * sizeof(float);
  const size_t SZ_H0H = (size_t)Bc * Tc * 64 * sizeof(_Float16);
  const size_t SZ_H1  = (size_t)Bc * Tc * 64 * sizeof(float);
  const size_t SZ_HP  = (size_t)Bc * 192 * sizeof(float);
  const size_t SZ_C2  = (size_t)Bc * 128 * 1025 * 9 * sizeof(_Float16);
  const size_t SZ_C3  = (size_t)Bc * 256 * 513 * 5 * sizeof(_Float16);
  const size_t SZ_C4  = (size_t)Bc * 64 * 257 * 3 * sizeof(_Float16);

  char* ws = (char*)d_ws;
  size_t o = 0;
  size_t o_xp  = o;            o = aup(o + SZ_XP);
  size_t o_xph = o;            o = aup(o + SZ_XPH);
  size_t o_xgA = o;            o = aup(o + SZ_XG);
  size_t o_xgB = o;            o = aup(o + SZ_XG);
  size_t o_h0h = o;            o = aup(o + SZ_H0H);
  size_t o_h1  = o;            o = aup(o + SZ_H1);
  size_t o_hp  = o;            o = aup(o + SZ_HP);
  size_t o_c1  = o_xgA;        // alias: xgA+xgB+h0h+h1 (176 MB) >= 143 MB, dead by conv time
  size_t o_c2  = o;            o = aup(o + SZ_C2);
  size_t o_c3  = o;            o = aup(o + SZ_C3);
  size_t o_c4  = o;            o = aup(o + SZ_C4);
  size_t o_cp  = o;            o = aup(o + (size_t)Bc * 64 * sizeof(float));

  float*     xp  = (float*)(ws + o_xp);
  _Float16*  xph = (_Float16*)(ws + o_xph);
  float*     xgA = (float*)(ws + o_xgA);
  float*     xgB = (float*)(ws + o_xgB);
  _Float16*  h0h = (_Float16*)(ws + o_h0h);
  float*     h1  = (float*)(ws + o_h1);
  float*     hp  = (float*)(ws + o_hp);
  _Float16*  c1  = (_Float16*)(ws + o_c1);
  _Float16*  c2  = (_Float16*)(ws + o_c2);
  _Float16*  c3  = (_Float16*)(ws + o_c3);
  _Float16*  c4  = (_Float16*)(ws + o_c4);
  float*     cp  = (float*)(ws + o_cp);

  // 1) FFN -> x_prim (f32 + f16)
  k_ffn<<<(Bc * Tc * Hc + 255) / 256, 256, 0, stream>>>(x, lengths, ffw, ffb, xp, xph);

  // 2) Layer-0 input-gate GEMMs (WMMA), then recurrence (both dirs concurrently)
  k_gates_gemm<32><<<256, 256, 0, stream>>>(xph, wih0f, bih0f, bhh0f, xgA);
  k_gates_gemm<32><<<256, 256, 0, stream>>>(xph, wih0r, bih0r, bhh0r, xgB);
  hipMemsetAsync(h0h, 0, SZ_H0H, stream);
  k_recur<<<2, 64, 0, stream>>>(xgA, xgB, whh0f, whh0r, lengths, (float*)nullptr, h0h);

  // 3) Layer-1 input-gate GEMMs (din=64, WMMA), then recurrence
  k_gates_gemm<64><<<256, 256, 0, stream>>>(h0h, wih1f, bih1f, bhh1f, xgA);
  k_gates_gemm<64><<<256, 256, 0, stream>>>(h0h, wih1r, bih1r, bhh1r, xgB);
  hipMemsetAsync(h1, 0, SZ_H1, stream);
  k_recur<<<2, 64, 0, stream>>>(xgA, xgB, whh1f, whh1r, lengths, h1, (_Float16*)nullptr);

  // 4) Sequence pooling (mean / max / last)
  k_pool<<<Bc, 64, 0, stream>>>(h1, lengths, hp);

  // 5) Conv stack: conv1 direct (CI=1, bandwidth-bound), conv2-4 WMMA implicit GEMM
  {
    long t1 = (long)Bc * 64 * 2049 * 17;
    k_conv1<<<(unsigned)((t1 + 255) / 256), 256, 0, stream>>>(xp, c1w, c1b, c1);
    int nt2 = (1025 * 9 + 15) / 16;   // 577
    k_conv_wmma<64><<<Bc * nt2, 256, 0, stream>>>(c1, c2w, c2b, c2, 2049, 17, 128, 1025, 9);
    int nt3 = (513 * 5 + 15) / 16;    // 161
    k_conv_wmma<128><<<Bc * nt3, 256, 0, stream>>>(c2, c3w, c3b, c3, 1025, 9, 256, 513, 5);
    int nt4 = (257 * 3 + 15) / 16;    // 49
    k_conv_wmma<256><<<Bc * nt4, 256, 0, stream>>>(c3, c4w, c4b, c4, 513, 5, 64, 257, 3);
  }
  k_cpool<<<(Bc * 64 + 255) / 256, 256, 0, stream>>>(c4, cp);

  // 6) Final linear -> (B, 4) f32
  k_final<<<1, 128, 0, stream>>>(cp, hp, lw, lb, (float*)d_out);
}
